// CrossAttentionBlock_14456859919004
// MI455X (gfx1250) — compile-verified, matrix operand order checked by reading, not run on hardware
//
#include <hip/hip_runtime.h>
#include <hip/hip_bf16.h>
#include <cstdint>

typedef __attribute__((ext_vector_type(16))) _Float16 v16h;
typedef __attribute__((ext_vector_type(8)))  _Float16 v8h;
typedef __attribute__((ext_vector_type(4)))  _Float16 v4h;
typedef __attribute__((ext_vector_type(8)))  float    v8f;
typedef __attribute__((ext_vector_type(4)))  float    v4f;

#define D_MODEL 1024
#define NH      16
#define HD      64
#define SEQ     2048
#define BATCH   2
#define M_ROWS  (BATCH*SEQ)          // 4096

#define ALPHA_C  0.5f
#define CLAMP_V  50.0f
#define LN_EPS   1e-5f

#define LDK 40   // 32-wide K tile + 8 halves pad (row stride 80B, 16B-aligned)
#define LDT 72   // 64-wide tile + 8 halves pad (row stride 144B, 16B-aligned)

// ---------------------------------------------------------------------------
// CDNA5 feature probes
// ---------------------------------------------------------------------------
#if defined(__has_builtin)
# if __has_builtin(__builtin_amdgcn_tensor_load_to_lds)
#  define HAVE_TDM 1
# endif
# if __has_builtin(__builtin_amdgcn_s_wait_tensorcnt)
#  define HAVE_TCNT 1
# endif
#endif
#ifndef HAVE_TDM
# define HAVE_TDM 0
#endif
#ifndef HAVE_TCNT
# define HAVE_TCNT 0
#endif

__device__ __forceinline__ void wait_tensorcnt0() {
#if HAVE_TCNT
  __builtin_amdgcn_s_wait_tensorcnt(0);
#elif HAVE_TDM
  asm volatile("s_wait_tensorcnt 0x0" ::: "memory");
#endif
}

#if HAVE_TDM
typedef __attribute__((ext_vector_type(4))) unsigned int u32x4;
typedef __attribute__((ext_vector_type(8))) int          i32x8;
typedef __attribute__((ext_vector_type(4))) int          i32x4;

// 1-D contiguous f16 TDM load: n_halves elements from gsrc into LDS at
// lds_byte_addr. If padded, the TDM inserts 4 DWORDs (8 halves) of pad after
// every 32 DWORDs (64 halves) stored -> matches the LDT=72 padded row layout.
__device__ __forceinline__ void tdm_load_f16(unsigned lds_byte_addr,
                                             const _Float16* gsrc,
                                             int n_halves, bool padded) {
  const unsigned long long ga = (unsigned long long)(uintptr_t)gsrc;
  u32x4 g0;
  g0[0] = 1u;                                     // count=1, user descriptor
  g0[1] = lds_byte_addr;                          // lds_addr
  g0[2] = (unsigned)(ga & 0xffffffffu);           // global_addr[31:0]
  g0[3] = (unsigned)((ga >> 32) & 0x01ffffffu)    // global_addr[56:32]
          | 0x80000000u;                          // type=2 ("image")
  i32x8 g1;
  unsigned w0 = (1u << 16);                       // data_size = 2 bytes
  if (padded)
    w0 |= (1u << 20)                              // pad_enable
        | (4u << 22)                              // pad_interval: 32 DWORDs
        | (3u << 25);                             // pad_amount:   4 DWORDs
  g1[0] = (int)w0;
  g1[1] = (int)(((unsigned)n_halves & 0xffffu) << 16); // tensor_dim0[15:0]
  g1[2] = (int)(((unsigned)n_halves >> 16)             // tensor_dim0[31:16]
          | (1u << 16));                               // tensor_dim1 = 1
  g1[3] = (int)(((unsigned)n_halves & 0xffffu) << 16); // tile_dim0
  g1[4] = 0;                                           // tile_dim1/2 unused
  g1[5] = n_halves;                                    // tensor_dim0_stride
  g1[6] = 0;
  g1[7] = 0;
  i32x4 z4 = {0, 0, 0, 0};
#if __clang_major__ >= 23
  i32x8 z8 = {0, 0, 0, 0, 0, 0, 0, 0};
  __builtin_amdgcn_tensor_load_to_lds(g0, g1, z4, z4, z8, 0);
#else
  __builtin_amdgcn_tensor_load_to_lds(g0, g1, z4, z4, 0);
#endif
}
#endif  // HAVE_TDM

__device__ __forceinline__ v8f wmma16x16x32(v16h a, v16h b, v8f c) {
  return __builtin_amdgcn_wmma_f32_16x16x32_f16(false, a, false, b, (short)0, c,
                                                false, false);
}

// 16x32 f16 fragment from LDS in the ISA A-matrix / B-transposed layout.
__device__ __forceinline__ v16h load_frag(const _Float16* base, int ld) {
  const int lane = threadIdx.x & 31;
  const _Float16* p = base + (lane & 15) * ld + ((lane >> 4) << 3);
  v16h f;
  ((v8h*)&f)[0] = *(const v8h*)p;
  ((v8h*)&f)[1] = *(const v8h*)(p + 16);
  return f;
}

// ---------------------------------------------------------------------------
// Q/K/V projection: out[b,h,pos,d](f16) = X @ W + bias. Block 128x128, K=32.
// Register-prefetch pipeline; packed f16 staging (b128 for A, b64 for B^T).
// ---------------------------------------------------------------------------
__global__ __launch_bounds__(256) void proj_kernel(
    const float* __restrict__ X, const float* __restrict__ W,
    const float* __restrict__ bias, _Float16* __restrict__ out) {
  __shared__ __align__(16) _Float16 As[128 * LDK];
  __shared__ __align__(16) _Float16 Bs[128 * LDK];   // stored [n][k]

  const int t    = threadIdx.x;
  const int m0   = blockIdx.y * 128;
  const int n0   = blockIdx.x * 128;
  const int w    = t >> 5;
  const int wm0  = (w >> 1) * 32;
  const int wn0  = (w & 1) * 64;
  const int lane = t & 31, half = lane >> 4, lcol = lane & 15;

  const int ar  = t >> 1,  ac = (t & 1) * 16;   // A: 2 thr/row, 16 f32 each
  const int bkq = t >> 5;                       // B: k-quad 0..7
  const int bn  = (t & 31) * 4;                 //    4 n values per thread

  v4f aR[4], bR[4];
  auto fetch = [&](int k0) {
    const float* ga = X + (size_t)(m0 + ar) * D_MODEL + k0 + ac;
    #pragma unroll
    for (int i = 0; i < 4; ++i) aR[i] = *(const v4f*)(ga + 4 * i);
    const float* gb = W + (size_t)(k0 + bkq * 4) * D_MODEL + n0 + bn;
    #pragma unroll
    for (int kk = 0; kk < 4; ++kk)
      bR[kk] = *(const v4f*)(gb + (size_t)kk * D_MODEL);
  };
  auto stage = [&]() {
    v8h p0, p1;
    #pragma unroll
    for (int i = 0; i < 4; ++i) {
      p0[i] = (_Float16)aR[0][i]; p0[4 + i] = (_Float16)aR[1][i];
      p1[i] = (_Float16)aR[2][i]; p1[4 + i] = (_Float16)aR[3][i];
    }
    *(v8h*)(As + ar * LDK + ac)     = p0;
    *(v8h*)(As + ar * LDK + ac + 8) = p1;
    #pragma unroll
    for (int nn = 0; nn < 4; ++nn) {
      v4h wv;
      #pragma unroll
      for (int kk = 0; kk < 4; ++kk) wv[kk] = (_Float16)bR[kk][nn];
      *(v4h*)(Bs + (bn + nn) * LDK + bkq * 4) = wv;   // transposed, b64 store
    }
  };

  v8f acc[2][4] = {};
  fetch(0); stage();
  __syncthreads();

  for (int k0 = 0; k0 < D_MODEL; k0 += 32) {
    const bool more = (k0 + 32) < D_MODEL;
    if (more) fetch(k0 + 32);     // global loads overlap the WMMAs below

    v16h af[2], bf[4];
    #pragma unroll
    for (int i = 0; i < 2; ++i) af[i] = load_frag(As + (wm0 + 16 * i) * LDK, LDK);
    #pragma unroll
    for (int j = 0; j < 4; ++j) bf[j] = load_frag(Bs + (wn0 + 16 * j) * LDK, LDK);
    #pragma unroll
    for (int i = 0; i < 2; ++i)
      #pragma unroll
      for (int j = 0; j < 4; ++j)
        acc[i][j] = wmma16x16x32(af[i], bf[j], acc[i][j]);

    __syncthreads();
    if (more) stage();
    __syncthreads();
  }

  #pragma unroll
  for (int i = 0; i < 2; ++i)
    #pragma unroll
    for (int j = 0; j < 4; ++j)
      #pragma unroll
      for (int r = 0; r < 8; ++r) {
        const int m = m0 + wm0 + 16 * i + r + 8 * half;
        const int n = n0 + wn0 + 16 * j + lcol;
        const int b = m >> 11, pos = m & (SEQ - 1);
        const int h = n >> 6,  d   = n & (HD - 1);
        const size_t idx = (((size_t)b * NH + h) * SEQ + pos) * HD + d;
        out[idx] = (_Float16)(acc[i][j][r] + bias[n]);
      }
}

// ---------------------------------------------------------------------------
// Flash attention per (b, h, 64-row q tile). 4 waves; Q/K tiles staged by the
// Tensor Data Mover (contiguous 8KB, padded into LDT rows); V transposed
// manually (TDM cannot transpose).
// ---------------------------------------------------------------------------
__global__ __launch_bounds__(128) void attn_kernel(
    const _Float16* __restrict__ Qp, const _Float16* __restrict__ Kp,
    const _Float16* __restrict__ Vp, const float* __restrict__ mask,
    _Float16* __restrict__ ctx) {
  __shared__ __align__(16) _Float16 Qs[64 * LDT];
  __shared__ __align__(16) _Float16 Ks[64 * LDT];   // [key][d] == B^T for Q.K^T
  __shared__ __align__(16) _Float16 Vs[64 * LDT];   // [d][key] == B^T for P.V
  __shared__ __align__(16) _Float16 Ps[64 * LDT];   // [q][key] == A for P.V

  const int t    = threadIdx.x;
  const int w    = t >> 5;
  const int lane = t & 31, half = lane >> 4, lcol = lane & 15;
  const int qt = blockIdx.x, h = blockIdx.y, b = blockIdx.z;

  const size_t headBase = ((size_t)b * NH + h) * SEQ * HD;
  const int q0 = qt * 64;
  const int lr = t >> 1, lc0 = (t & 1) * 32;

  // ---- Q tile (resident) ----
#if HAVE_TDM
  if (w == 0)
    tdm_load_f16((unsigned)(uintptr_t)Qs, Qp + headBase + (size_t)q0 * HD,
                 64 * HD, /*padded=*/true);
  wait_tensorcnt0();
#else
  {
    const _Float16* g = Qp + headBase + (size_t)(q0 + lr) * HD + lc0;
    _Float16* l = Qs + lr * LDT + lc0;
    #pragma unroll
    for (int i = 0; i < 32; i += 8) *(v8h*)(l + i) = *(const v8h*)(g + i);
  }
#endif

  v8f o[4] = {};
  float mrow[8], lrow[8];
  #pragma unroll
  for (int r = 0; r < 8; ++r) { mrow[r] = -3.0e38f; lrow[r] = 0.f; }
  const float scale = 0.125f;   // 1/sqrt(hd)

  for (int j0 = 0; j0 < SEQ; j0 += 64) {
    __syncthreads();
    // ---- K tile (TDM) + V tile transposed (manual) ----
#if HAVE_TDM
    if (w == 0)
      tdm_load_f16((unsigned)(uintptr_t)Ks, Kp + headBase + (size_t)j0 * HD,
                   64 * HD, /*padded=*/true);
#else
    {
      const _Float16* gk = Kp + headBase + (size_t)(j0 + lr) * HD + lc0;
      _Float16* lk = Ks + lr * LDT + lc0;
      #pragma unroll
      for (int i = 0; i < 32; i += 8) *(v8h*)(lk + i) = *(const v8h*)(gk + i);
    }
#endif
    {
      const _Float16* gv = Vp + headBase + (size_t)(j0 + lr) * HD + lc0;
      #pragma unroll
      for (int i = 0; i < 32; ++i) Vs[(lc0 + i) * LDT + lr] = gv[i];
    }
#if HAVE_TDM
    wait_tensorcnt0();
#endif
    __syncthreads();

    // ---- S = Q @ K^T (16 x 64 per wave) ----
    v8f s[4] = {};
    #pragma unroll
    for (int k0 = 0; k0 < HD; k0 += 32) {
      v16h aq = load_frag(Qs + (w * 16) * LDT + k0, LDT);
      #pragma unroll
      for (int j4 = 0; j4 < 4; ++j4) {
        v16h bkf = load_frag(Ks + (j4 * 16) * LDT + k0, LDT);
        s[j4] = wmma16x16x32(aq, bkf, s[j4]);
      }
    }

    // ---- mask blend + clamp + online softmax ----
    #pragma unroll
    for (int r = 0; r < 8; ++r) {
      const int qg = q0 + w * 16 + r + 8 * half;
      float sv[4];
      float rm = -3.0e38f;
      #pragma unroll
      for (int j4 = 0; j4 < 4; ++j4) {
        const int kg = j0 + j4 * 16 + lcol;
        float x = s[j4][r] * scale;
        const float mk = mask[((size_t)b * SEQ + qg) * SEQ + kg];
        x = x * (ALPHA_C * mk + (1.0f - ALPHA_C));
        x = fminf(fmaxf(x, -CLAMP_V), CLAMP_V);
        sv[j4] = x;
        rm = fmaxf(rm, x);
      }
      #pragma unroll
      for (int off = 1; off < 16; off <<= 1)
        rm = fmaxf(rm, __shfl_xor(rm, off, 32));
      const float mnew = fmaxf(mrow[r], rm);
      const float sc = __expf(mrow[r] - mnew);
      float rs = 0.f;
      #pragma unroll
      for (int j4 = 0; j4 < 4; ++j4) {
        const float p = __expf(sv[j4] - mnew);
        rs += p;
        Ps[(w * 16 + r + 8 * half) * LDT + j4 * 16 + lcol] = (_Float16)p;
      }
      #pragma unroll
      for (int off = 1; off < 16; off <<= 1)
        rs += __shfl_xor(rs, off, 32);
      lrow[r] = lrow[r] * sc + rs;
      mrow[r] = mnew;
      #pragma unroll
      for (int j4 = 0; j4 < 4; ++j4) o[j4][r] *= sc;
    }
    __syncthreads();

    // ---- O += P @ V ----
    #pragma unroll
    for (int k0 = 0; k0 < 64; k0 += 32) {
      v16h ap = load_frag(Ps + (w * 16) * LDT + k0, LDT);
      #pragma unroll
      for (int j4 = 0; j4 < 4; ++j4) {
        v16h bvf = load_frag(Vs + (j4 * 16) * LDT + k0, LDT);
        o[j4] = wmma16x16x32(ap, bvf, o[j4]);
      }
    }
  }

  // normalize, emit ctx f16 [B,Q,D]
  #pragma unroll
  for (int r = 0; r < 8; ++r) {
    const int qg = q0 + w * 16 + r + 8 * half;
    const float inv = 1.0f / lrow[r];
    #pragma unroll
    for (int j4 = 0; j4 < 4; ++j4) {
      const int d = j4 * 16 + lcol;
      ctx[((size_t)b * SEQ + qg) * D_MODEL + h * HD + d] =
          (_Float16)(o[j4][r] * inv);
    }
  }
}

// ---------------------------------------------------------------------------
// Output projection + bias + residual: resid = ctx(f16) @ Wo + bo + query
// ---------------------------------------------------------------------------
__global__ __launch_bounds__(256) void outproj_kernel(
    const _Float16* __restrict__ ctx, const float* __restrict__ Wo,
    const float* __restrict__ bo, const float* __restrict__ query,
    float* __restrict__ resid) {
  __shared__ __align__(16) _Float16 As[128 * LDK];
  __shared__ __align__(16) _Float16 Bs[128 * LDK];

  const int t    = threadIdx.x;
  const int m0   = blockIdx.y * 128;
  const int n0   = blockIdx.x * 128;
  const int w    = t >> 5;
  const int wm0  = (w >> 1) * 32;
  const int wn0  = (w & 1) * 64;
  const int lane = t & 31, half = lane >> 4, lcol = lane & 15;
  const int ar  = t >> 1,  ac = (t & 1) * 16;
  const int bkq = t >> 5;
  const int bn  = (t & 31) * 4;

  v8h aH[2];
  v4f bR[4];
  auto fetch = [&](int k0) {
    const _Float16* ga = ctx + (size_t)(m0 + ar) * D_MODEL + k0 + ac;
    aH[0] = *(const v8h*)(ga);
    aH[1] = *(const v8h*)(ga + 8);
    const float* gb = Wo + (size_t)(k0 + bkq * 4) * D_MODEL + n0 + bn;
    #pragma unroll
    for (int kk = 0; kk < 4; ++kk)
      bR[kk] = *(const v4f*)(gb + (size_t)kk * D_MODEL);
  };
  auto stage = [&]() {
    *(v8h*)(As + ar * LDK + ac)     = aH[0];
    *(v8h*)(As + ar * LDK + ac + 8) = aH[1];
    #pragma unroll
    for (int nn = 0; nn < 4; ++nn) {
      v4h wv;
      #pragma unroll
      for (int kk = 0; kk < 4; ++kk) wv[kk] = (_Float16)bR[kk][nn];
      *(v4h*)(Bs + (bn + nn) * LDK + bkq * 4) = wv;
    }
  };

  v8f acc[2][4] = {};
  fetch(0); stage();
  __syncthreads();

  for (int k0 = 0; k0 < D_MODEL; k0 += 32) {
    const bool more = (k0 + 32) < D_MODEL;
    if (more) fetch(k0 + 32);

    v16h af[2], bf[4];
    #pragma unroll
    for (int i = 0; i < 2; ++i) af[i] = load_frag(As + (wm0 + 16 * i) * LDK, LDK);
    #pragma unroll
    for (int j = 0; j < 4; ++j) bf[j] = load_frag(Bs + (wn0 + 16 * j) * LDK, LDK);
    #pragma unroll
    for (int i = 0; i < 2; ++i)
      #pragma unroll
      for (int j = 0; j < 4; ++j)
        acc[i][j] = wmma16x16x32(af[i], bf[j], acc[i][j]);

    __syncthreads();
    if (more) stage();
    __syncthreads();
  }

  #pragma unroll
  for (int i = 0; i < 2; ++i)
    #pragma unroll
    for (int j = 0; j < 4; ++j)
      #pragma unroll
      for (int r = 0; r < 8; ++r) {
        const int m = m0 + wm0 + 16 * i + r + 8 * half;
        const int n = n0 + wn0 + 16 * j + lcol;
        const size_t idx = (size_t)m * D_MODEL + n;
        resid[idx] = acc[i][j][r] + bo[n] + query[idx];
      }
}

// ---------------------------------------------------------------------------
// Row-wise LayerNorm: out = (x - mu) * rsqrt(var + eps) * gamma + beta
// ---------------------------------------------------------------------------
__global__ __launch_bounds__(256) void ln_kernel(
    const float* __restrict__ resid, const float* __restrict__ gamma,
    const float* __restrict__ beta, float* __restrict__ out) {
  __shared__ float red[256];
  const int row = blockIdx.x;
  const float* x = resid + (size_t)row * D_MODEL;

  float s = 0.f, s2 = 0.f;
  for (int c = threadIdx.x; c < D_MODEL; c += 256) {
    const float v = x[c];
    s += v; s2 += v * v;
  }
  red[threadIdx.x] = s; __syncthreads();
  for (int off = 128; off > 0; off >>= 1) {
    if (threadIdx.x < off) red[threadIdx.x] += red[threadIdx.x + off];
    __syncthreads();
  }
  const float mean = red[0] * (1.0f / D_MODEL);
  __syncthreads();
  red[threadIdx.x] = s2; __syncthreads();
  for (int off = 128; off > 0; off >>= 1) {
    if (threadIdx.x < off) red[threadIdx.x] += red[threadIdx.x + off];
    __syncthreads();
  }
  const float var = red[0] * (1.0f / D_MODEL) - mean * mean;
  const float inv = rsqrtf(var + LN_EPS);
  for (int c = threadIdx.x; c < D_MODEL; c += 256)
    out[(size_t)row * D_MODEL + c] = (x[c] - mean) * inv * gamma[c] + beta[c];
}

// ---------------------------------------------------------------------------
extern "C" void kernel_launch(void* const* d_in, const int* in_sizes, int n_in,
                              void* d_out, int out_size, void* d_ws,
                              size_t ws_size, hipStream_t stream) {
  const float* query  = (const float*)d_in[0];
  const float* key_in = (const float*)d_in[1];
  const float* mask   = (const float*)d_in[2];
  const float* Wq = (const float*)d_in[3];
  const float* bq = (const float*)d_in[4];
  const float* Wk = (const float*)d_in[5];
  const float* bk = (const float*)d_in[6];
  const float* Wv = (const float*)d_in[7];
  const float* bv = (const float*)d_in[8];
  const float* Wo = (const float*)d_in[9];
  const float* bo = (const float*)d_in[10];
  const float* gamma = (const float*)d_in[11];
  const float* beta  = (const float*)d_in[12];

  // Workspace: 3 x 8MB f16 head tensors + 8MB f16 ctx + 16MB f32 resid = 48MB
  const size_t HEADEL = (size_t)BATCH * NH * SEQ * HD;   // 4M elements
  _Float16* Qp  = (_Float16*)d_ws;
  _Float16* Kp  = Qp + HEADEL;
  _Float16* Vp  = Kp + HEADEL;
  _Float16* ctx = Vp + HEADEL;
  float* resid  = (float*)(ctx + HEADEL);

  dim3 pgrid(D_MODEL / 128, M_ROWS / 128);   // (8, 32)
  proj_kernel<<<pgrid, 256, 0, stream>>>(query,  Wq, bq, Qp);
  proj_kernel<<<pgrid, 256, 0, stream>>>(key_in, Wk, bk, Kp);
  proj_kernel<<<pgrid, 256, 0, stream>>>(key_in, Wv, bv, Vp);

  dim3 agrid(SEQ / 64, NH, BATCH);
  attn_kernel<<<agrid, 128, 0, stream>>>(Qp, Kp, Vp, mask, ctx);

  outproj_kernel<<<pgrid, 256, 0, stream>>>(ctx, Wo, bo, query, resid);

  ln_kernel<<<M_ROWS, 256, 0, stream>>>(resid, gamma, beta, (float*)d_out);
}
